// DeepONetGradNet_68607807586373
// MI455X (gfx1250) — compile-verified, matrix operand order checked by reading, not run on hardware
//
#include <hip/hip_runtime.h>
#include <hip/hip_bf16.h>
#include <math.h>

// ---------------------------------------------------------------------------
// Problem constants (from reference): N=32768, NX=64, K=16, E=32, DY=32
// ---------------------------------------------------------------------------
#define NN  32768
#define NX  64
#define KK  16
#define EE  32
#define DY  32
#define KE  512          // K*E, the contraction depth
#define JC  4096         // NX*NX, jacobian columns per sample
#define KH  256          // K elements per double-buffer stage in jac kernel

typedef _Float16 half_t;
typedef __attribute__((ext_vector_type(16))) _Float16 v16h;
typedef __attribute__((ext_vector_type(8)))  _Float16 v8h;
typedef __attribute__((ext_vector_type(8)))  float    v8f;

// ---------------------------------------------------------------------------
// WMMA f16 fragment loader (wave32, V_WMMA_F32_16X16X32_F16 layouts).
// A (16xK=32, row-major src, per-lane row = lane&15):
//   lanes 0-15 : K = 0..7   (v0..v3), K = 16..23 (v4..v7)
//   lanes 16-31: K = 8..15  (v0..v3), K = 24..31 (v4..v7)
// B (K=32 x 16) mirrors this with "row" -> output column, so storing B
// transposed (col-major as rows of K) lets one loader serve both operands.
// Caller passes p = base + row*ld + k0 + (lane>=16 ? 8 : 0).
// ---------------------------------------------------------------------------
__device__ __forceinline__ v16h load_frag(const half_t* p) {
  v8h lo = *(const v8h*)(p);        // K = kb .. kb+7
  v8h hi = *(const v8h*)(p + 16);   // K = kb+16 .. kb+23
  v16h r;
#pragma unroll
  for (int t = 0; t < 8; ++t) { r[t] = lo[t]; r[8 + t] = hi[t]; }
  return r;
}

__device__ __forceinline__ v8f wmma_f16(v16h a, v16h b, v8f c) {
  return __builtin_amdgcn_wmma_f32_16x16x32_f16(
      /*neg_a=*/false, a, /*neg_b=*/false, b,
      /*c_mod=*/(short)0, c, /*reuse_a=*/false, /*reuse_b=*/false);
}

// ---------------------------------------------------------------------------
// gfx1250 async global->LDS copy (16B per lane), tracked by ASYNCcnt.
// LDS destination operand is the LDS byte offset: for generic shared
// pointers the shared aperture occupies the high DWORD, so addr[31:0] is
// exactly the LDS offset.
// ---------------------------------------------------------------------------
__device__ __forceinline__ void async_copy_b128(half_t* lds_dst, const half_t* gsrc) {
  unsigned lds_off = (unsigned)(size_t)(void*)lds_dst;
  asm volatile("global_load_async_to_lds_b128 %0, %1, off"
               :: "v"(lds_off), "v"(gsrc)
               : "memory");
}

__device__ __forceinline__ void wait_async_all() {
  asm volatile("s_wait_asynccnt 0x0" ::: "memory");
}

// Branch-free tanh: 1 - 2*rcp(exp(2x)+1); saturates correctly at +-inf.
__device__ __forceinline__ float fast_tanh(float x) {
  float t = __expf(2.0f * x);
  return 1.0f - 2.0f * __builtin_amdgcn_rcpf(t + 1.0f);
}

// ---------------------------------------------------------------------------
// Pack kernels: fp32 -> f16 operand layouts
// ---------------------------------------------------------------------------
__global__ void pack_x_kernel(const float* __restrict__ x, half_t* __restrict__ Xh) {
  int i = blockIdx.x * 256 + threadIdx.x;      // N*NX = 2M elements
  Xh[i] = (half_t)x[i];
}

// W1T[ke][i] = W1[k][i][e]   (512 x 64)   -- B operand of trunk GEMM
// W2T[o][ke] = W2[k][e][o]   (64 x 512)   -- B operand of z GEMM
__global__ void pack_w_kernel(const float* __restrict__ W1, const float* __restrict__ W2,
                              half_t* __restrict__ W1T, half_t* __restrict__ W2T) {
  int idx = blockIdx.x * 256 + threadIdx.x;    // 32768 elements each
  {
    int ke = idx >> 6, i = idx & 63;
    int k = ke >> 5, e = ke & 31;
    W1T[idx] = (half_t)W1[k * (NX * EE) + i * EE + e];
  }
  {
    int o = idx >> 9, ke = idx & 511;
    int k = ke >> 5, e = ke & 31;
    W2T[idx] = (half_t)W2[k * (EE * NX) + e * NX + o];
  }
}

// W12T[o*64+i][ke] = W1[k][i][e] * W2[k][e][o]   (4096 x 512, 4 MB f16)
__global__ void pack_w12_kernel(const float* __restrict__ W1, const float* __restrict__ W2,
                                half_t* __restrict__ W12T) {
  size_t idx = (size_t)blockIdx.x * 256 + threadIdx.x;   // 4096*512 = 2M
  int c  = (int)(idx >> 9);            // c = o*64 + i
  int ke = (int)(idx & 511);
  int k = ke >> 5, e = ke & 31;
  int o = c >> 6, i = c & 63;
  W12T[idx] = (half_t)(W1[k * (NX * EE) + i * EE + e] *
                       W2[k * (EE * NX) + e * NX + o]);
}

// ---------------------------------------------------------------------------
// Per-sample small nets (fp32 VALU; weights staged in LDS).
// Produces: wgt[n][k] = softplus(relu(y_n@Wb1+bb1)@Wb2+bb2)
//           bias_tot[n][o] = relu(y_n@Wc1+bc1)@Wc2 + bc2 + wgt@b2
// ---------------------------------------------------------------------------
__device__ __forceinline__ float softplus_f(float s) {
  return (s > 20.0f) ? s : log1pf(expf(s));
}

__global__ __launch_bounds__(256)
void smallnet_kernel(const float* __restrict__ y,
                     const float* __restrict__ Wb1, const float* __restrict__ bb1,
                     const float* __restrict__ Wb2, const float* __restrict__ bb2,
                     const float* __restrict__ Wc1, const float* __restrict__ bc1,
                     const float* __restrict__ Wc2, const float* __restrict__ bc2,
                     const float* __restrict__ b2,
                     const float* __restrict__ mean, const float* __restrict__ var,
                     float* __restrict__ wgt, float* __restrict__ bias_tot) {
  __shared__ float sWb1[DY * 64], sWb2[64 * KK], sWc1[DY * 16], sWc2[16 * NX], sB2[KK * NX];
  __shared__ float sbb1[64], sbc2[NX], sbb2[KK], sbc1[16], sMean[DY], sVinv[DY];
  int tid = threadIdx.x;
  for (int i = tid; i < DY * 64; i += 256) sWb1[i] = Wb1[i];
  for (int i = tid; i < 64 * KK; i += 256) sWb2[i] = Wb2[i];
  for (int i = tid; i < DY * 16; i += 256) sWc1[i] = Wc1[i];
  for (int i = tid; i < 16 * NX; i += 256) sWc2[i] = Wc2[i];
  for (int i = tid; i < KK * NX; i += 256) sB2[i]  = b2[i];
  if (tid < 64) { sbb1[tid] = bb1[tid]; sbc2[tid] = bc2[tid]; }
  if (tid < DY) { sMean[tid] = mean[tid]; sVinv[tid] = rsqrtf(var[tid]); }
  if (tid < KK) { sbb2[tid] = bb2[tid]; sbc1[tid] = bc1[tid]; }
  __syncthreads();

  int n = blockIdx.x * 256 + tid;

  float yn[DY];
#pragma unroll
  for (int d = 0; d < DY; ++d) {
    float v = (y[(size_t)n * DY + d] - sMean[d]) * sVinv[d];
    yn[d] = fminf(fmaxf(v, -10.0f), 10.0f);
  }

  float t1[64];
#pragma unroll 4
  for (int j = 0; j < 64; ++j) {
    float s = sbb1[j];
#pragma unroll
    for (int d = 0; d < DY; ++d) s = fmaf(yn[d], sWb1[d * 64 + j], s);
    t1[j] = fmaxf(s, 0.0f);
  }

  float w[KK];
#pragma unroll
  for (int k = 0; k < KK; ++k) {
    float s = sbb2[k];
#pragma unroll 8
    for (int j = 0; j < 64; ++j) s = fmaf(t1[j], sWb2[j * KK + k], s);
    w[k] = softplus_f(s);
    wgt[(size_t)n * KK + k] = w[k];
  }

  float t2[16];
#pragma unroll
  for (int q = 0; q < 16; ++q) {
    float s = sbc1[q];
#pragma unroll
    for (int d = 0; d < DY; ++d) s = fmaf(yn[d], sWc1[d * 16 + q], s);
    t2[q] = fmaxf(s, 0.0f);
  }

#pragma unroll 4
  for (int o = 0; o < NX; ++o) {
    float s = sbc2[o];
#pragma unroll
    for (int q = 0; q < 16; ++q) s = fmaf(t2[q], sWc2[q * NX + o], s);
#pragma unroll
    for (int k = 0; k < KK; ++k) s = fmaf(w[k], sB2[k * NX + o], s);
    bias_tot[(size_t)n * NX + o] = s;
  }
}

// ---------------------------------------------------------------------------
// Trunk GEMM: pre[n][ke] = Xh(N x 64) @ W1T^T + b1  via WMMA f16 -> f32
// fused: h = tanh(pre); M = w*(1-h^2); A2 = w*h  (both f16, row-major N x 512)
// Grid (N/16, 8), 128 threads (4 waves); each wave one 16x16 tile.
// ---------------------------------------------------------------------------
__global__ __launch_bounds__(128)
void trunk_kernel(const half_t* __restrict__ Xh, const half_t* __restrict__ W1T,
                  const float* __restrict__ b1, const float* __restrict__ wgt,
                  half_t* __restrict__ Mmat, half_t* __restrict__ A2mat) {
  int lane = threadIdx.x & 31, wave = threadIdx.x >> 5;
  int r0 = blockIdx.x * 16;                     // sample tile base
  int c0 = (blockIdx.y * 4 + wave) * 16;        // ke column tile base
  int mrow = lane & 15;
  int half_id = lane >> 4;
  int koff = half_id * 8;
  int colke = c0 + mrow;                        // this lane's D column

  float bcol = b1[colke];                       // b1 flat (K,E) == ke index
  v8f acc = {bcol, bcol, bcol, bcol, bcol, bcol, bcol, bcol};

#pragma unroll
  for (int ks = 0; ks < 2; ++ks) {              // K = 64 in steps of 32
    int kb = ks * 32 + koff;
    v16h a = load_frag(Xh  + (size_t)(r0 + mrow) * NX + kb);
    v16h b = load_frag(W1T + (size_t)colke      * NX + kb);
    acc = wmma_f16(a, b, acc);
  }

  int kidx = colke >> 5;                        // which expert k
#pragma unroll
  for (int v = 0; v < 8; ++v) {
    int n = r0 + v + half_id * 8;
    float h  = fast_tanh(acc[v]);
    float wk = wgt[(size_t)n * KK + kidx];
    Mmat[(size_t)n * KE + colke]  = (half_t)(wk * (1.0f - h * h));
    A2mat[(size_t)n * KE + colke] = (half_t)(wk * h);
  }
}

// ---------------------------------------------------------------------------
// z GEMM: zdelta = A2(N x 512) @ W2T^T ; z = x + zdelta + bias_tot
// Grid N/16, 128 threads; wave w covers output cols [16w, 16w+16).
// ---------------------------------------------------------------------------
__global__ __launch_bounds__(128)
void z_kernel(const half_t* __restrict__ A2mat, const half_t* __restrict__ W2T,
              const float* __restrict__ x, const float* __restrict__ bias_tot,
              float* __restrict__ z_out) {
  int lane = threadIdx.x & 31, wave = threadIdx.x >> 5;
  int r0 = blockIdx.x * 16;
  int mrow = lane & 15;
  int half_id = lane >> 4;
  int koff = half_id * 8;
  int col = wave * 16 + mrow;

  v8f acc = {};
#pragma unroll
  for (int ks = 0; ks < 16; ++ks) {             // K = 512
    int kb = ks * 32 + koff;
    v16h a = load_frag(A2mat + (size_t)(r0 + mrow) * KE + kb);
    v16h b = load_frag(W2T   + (size_t)col        * KE + kb);
    acc = wmma_f16(a, b, acc);
  }

#pragma unroll
  for (int v = 0; v < 8; ++v) {
    int n = r0 + v + half_id * 8;
    size_t idx = (size_t)n * NX + col;
    z_out[idx] = x[idx] + bias_tot[idx] + acc[v];
  }
}

// ---------------------------------------------------------------------------
// Jacobian GEMM: jac = I + M(N x 512) @ W12T^T(512 x 4096)
// Block: 4 waves, 64x64 output tile. K split into 2 halves of 256;
// each half's A/B strips (64x256 f16 each) staged via async global->LDS
// loads (ASYNCcnt) with double buffering: half 1 loads overlap half 0 WMMAs.
// 128 KB LDS total -> 2 blocks per WGP.
// ---------------------------------------------------------------------------
__global__ __launch_bounds__(128)
void jac_kernel(const half_t* __restrict__ Mmat, const half_t* __restrict__ W12T,
                float* __restrict__ jac_out) {
  extern __shared__ half_t lds[];   // [2 bufs][A 64xKH | B 64xKH]

  int tid = threadIdx.x;            // 128 threads
  int r0 = blockIdx.x * 64;         // sample base
  int c0 = blockIdx.y * 64;         // jac column base

  int lane = tid & 31, wave = tid >> 5;
  int mrow = lane & 15;
  int half_id = lane >> 4;
  int koff = half_id * 8;
  int arow = wave * 16 + mrow;

  // Issue async stage of K-half `kh` into buffer `buf`.
  auto stage = [&](int buf, int kh) {
    half_t* As = lds + buf * (2 * 64 * KH);
    half_t* Bs = As + 64 * KH;
    int kbase = kh * KH;
    for (int idx = tid; idx < 64 * (KH / 8); idx += 128) {  // 64 rows x 32 segs
      int row = idx >> 5, seg = (idx & 31) * 8;
      async_copy_b128(As + row * KH + seg,
                      Mmat + (size_t)(r0 + row) * KE + kbase + seg);
      async_copy_b128(Bs + row * KH + seg,
                      W12T + (size_t)(c0 + row) * KE + kbase + seg);
    }
  };

  v8f acc[4] = {};

  stage(0, 0);
  wait_async_all();
  __syncthreads();

  stage(1, 1);                      // overlap with compute on buffer 0

#pragma unroll
  for (int buf = 0; buf < 2; ++buf) {
    const half_t* As = lds + buf * (2 * 64 * KH);
    const half_t* Bs = As + 64 * KH;
#pragma unroll
    for (int ks = 0; ks < KH / 32; ++ks) {      // 8 WMMA K-steps per half
      int kb = ks * 32 + koff;
      v16h a = load_frag(As + arow * KH + kb);
#pragma unroll
      for (int c = 0; c < 4; ++c) {
        v16h b = load_frag(Bs + (c * 16 + mrow) * KH + kb);
        acc[c] = wmma_f16(a, b, acc[c]);
      }
    }
    if (buf == 0) {                 // make buffer 1 visible before use
      wait_async_all();
      __syncthreads();
    }
  }

#pragma unroll
  for (int c = 0; c < 4; ++c) {
    int col = c0 + c * 16 + mrow;             // col = o*64 + i
    float eye = ((col >> 6) == (col & 63)) ? 1.0f : 0.0f;
#pragma unroll
    for (int v = 0; v < 8; ++v) {
      int n = r0 + wave * 16 + v + half_id * 8;
      jac_out[(size_t)n * JC + col] = acc[c][v] + eye;
    }
  }
}

// ---------------------------------------------------------------------------
// Launch
// ---------------------------------------------------------------------------
extern "C" void kernel_launch(void* const* d_in, const int* in_sizes, int n_in,
                              void* d_out, int out_size, void* d_ws, size_t ws_size,
                              hipStream_t stream) {
  const float* x    = (const float*)d_in[0];
  const float* y    = (const float*)d_in[1];
  const float* W1   = (const float*)d_in[2];
  const float* b1   = (const float*)d_in[3];
  const float* W2   = (const float*)d_in[4];
  const float* b2   = (const float*)d_in[5];
  const float* Wb1  = (const float*)d_in[6];
  const float* bb1  = (const float*)d_in[7];
  const float* Wb2  = (const float*)d_in[8];
  const float* bb2  = (const float*)d_in[9];
  const float* Wc1  = (const float*)d_in[10];
  const float* bc1  = (const float*)d_in[11];
  const float* Wc2  = (const float*)d_in[12];
  const float* bc2  = (const float*)d_in[13];
  const float* mean = (const float*)d_in[14];
  const float* var  = (const float*)d_in[15];

  // Workspace layout (256B-aligned offsets), ~86 MB total.
  char* ws = (char*)d_ws;
  half_t* Xh       = (half_t*)(ws + 0);                       // 4 MB
  half_t* W1T      = (half_t*)(ws + 4194304);                 // 64 KB
  half_t* W2T      = (half_t*)(ws + 4259840);                 // 64 KB
  half_t* W12T     = (half_t*)(ws + 4325376);                 // 4 MB
  float*  wgt      = (float*) (ws + 8519680);                 // 2 MB
  float*  bias_tot = (float*) (ws + 10616832);                // 8 MB
  half_t* Mmat     = (half_t*)(ws + 19005440);                // 32 MB
  half_t* A2mat    = (half_t*)(ws + 52559872);                // 32 MB

  float* z_out   = (float*)d_out;                 // (N, NX)
  float* jac_out = z_out + (size_t)NN * NX;       // (N, NX, NX)

  pack_x_kernel  <<<(NN * NX) / 256, 256, 0, stream>>>(x, Xh);
  pack_w_kernel  <<<(KE * NX) / 256, 256, 0, stream>>>(W1, W2, W1T, W2T);
  pack_w12_kernel<<<(JC * KE) / 256, 256, 0, stream>>>(W1, W2, W12T);

  smallnet_kernel<<<NN / 256, 256, 0, stream>>>(
      y, Wb1, bb1, Wb2, bb2, Wc1, bc1, Wc2, bc2, b2, mean, var, wgt, bias_tot);

  trunk_kernel<<<dim3(NN / 16, 8), 128, 0, stream>>>(Xh, W1T, b1, wgt, Mmat, A2mat);

  z_kernel<<<NN / 16, 128, 0, stream>>>(A2mat, W2T, x, bias_tot, z_out);

  jac_kernel<<<dim3(NN / 64, JC / 64), 128, 4 * 64 * KH * sizeof(half_t), stream>>>(
      Mmat, W12T, jac_out);
}
